// GRU_30142080483560
// MI455X (gfx1250) — matrix operand order repaired and checked by reading, not compiled, not run on hardware
//
#include <hip/hip_runtime.h>
#include <hip/hip_bf16.h>
#include <cstdint>
#include <cstddef>

#define B_  64
#define T_  512
#define V_  4096
#define H_  1024
#define KC  64              // K-chunk staged in LDS per stage
#define LDP 72              // padded LDS row stride (elements) to dodge bank conflicts
#define NSTAGE (H_ / KC)    // 16

typedef __bf16 v8bf  __attribute__((ext_vector_type(8)));
typedef __bf16 v16bf __attribute__((ext_vector_type(16)));
typedef float  v8f   __attribute__((ext_vector_type(8)));

__device__ __forceinline__ v16bf bfcat(v8bf a, v8bf b) {
  return __builtin_shufflevector(a, b, 0,1,2,3,4,5,6,7,8,9,10,11,12,13,14,15);
}

__device__ __forceinline__ v16bf loadB(const __bf16* p) {
  return bfcat(*(const v8bf*)p, *(const v8bf*)(p + 8));
}

// Asynchronously copy a [64 x KC] bf16 slice (cols kc..kc+KC) of A[64][H_] into LDS.
// 64*64*2B = 8 KiB = 512 x 16B chunks; 256 threads x 2 chunks each.
// CDNA5 async-to-LDS path (tracked by ASYNCcnt, waited with s_wait_asynccnt).
__device__ __forceinline__ void async_stage(const __bf16* __restrict__ A, int kc,
                                            __bf16* dst /* LDS, row stride LDP */) {
  const int tid = threadIdx.x;
#pragma unroll
  for (int j = 0; j < 2; ++j) {
    const int c   = tid + 256 * j;
    const int row = c >> 3;
    const int c16 = c & 7;
    const uint64_t gaddr = (uint64_t)(size_t)(A + (size_t)row * H_ + kc + c16 * 8);
    const unsigned lds   = (unsigned)(size_t)(const void*)(dst + row * LDP + c16 * 8);
    asm volatile("global_load_async_to_lds_b128 %0, %1, off"
                 :: "v"(lds), "v"(gaddr)
                 : "memory");
  }
}

// Staged GEMM: C[64 x (NT*16)] += A[64 x H_] * W[H_ x (NT*16)],
// A bf16 row-major in global, W given transposed as Wt[N][H_] bf16 row-major.
// Each wave owns one 16-row strip (mt) and NT 16-col tiles starting at colBase.
// Double-buffered async LDS staging of A; B fragments software-pipelined in
// registers one K-chunk ahead. B base pointers are loop-invariant (address-
// space inference keeps them on the GLOBAL path); only a uniform scalar
// element offset `koff` rolls forward (SGPR, GVS addressing).
template <int NT>
__device__ __forceinline__ void gemm_staged(
    const __bf16* __restrict__ A,
    const __bf16* __restrict__ Wt,
    int mt, int colBase,
    __bf16* As,               // LDS: 2 * 64 * LDP elements
    v8f acc[NT])
{
  const int lane = threadIdx.x & 31;
  const int kh   = lane >> 4;   // K-half select (A and B: +16 group)
  const int nl   = lane & 15;   // A row within strip / B output column

  const __bf16* bbase[NT];      // loop-invariant, provably global
#pragma unroll
  for (int tt = 0; tt < NT; ++tt)
    bbase[tt] = Wt + (size_t)(colBase + tt * 16 + nl) * H_ + kh * 16;

  v16bf bcur[NT], bnxt[NT];

  async_stage(A, 0, As);
#pragma unroll
  for (int tt = 0; tt < NT; ++tt) bcur[tt] = loadB(bbase[tt]);  // stage 0, k32=0

  int koff = 0;                 // uniform K element offset (SGPR)
#pragma unroll 2
  for (int s = 0; s < NSTAGE; ++s) {
    if (s + 1 < NSTAGE) {
      async_stage(A, (s + 1) * KC, As + ((s + 1) & 1) * (64 * LDP));
      asm volatile("s_wait_asynccnt 0x2" ::: "memory");  // stage s ready (in-order)
    } else {
      asm volatile("s_wait_asynccnt 0x0" ::: "memory");
    }
    __syncthreads();
    const __bf16* Ab = As + (s & 1) * (64 * LDP) + (mt * 16 + nl) * LDP;

    // ---- K chunk k32 = 0 (consumes bcur, issues bnxt) ----
    v16bf a0 = bfcat(*(const v8bf*)(Ab + kh * 16),
                     *(const v8bf*)(Ab + 8 + kh * 16));
#pragma unroll
    for (int tt = 0; tt < NT; ++tt) bnxt[tt] = loadB(bbase[tt] + koff + 32);
#pragma unroll
    for (int tt = 0; tt < NT; ++tt)
      acc[tt] = __builtin_amdgcn_wmma_f32_16x16x32_bf16(
          false, a0, false, bcur[tt], (short)0, acc[tt], false, false);

    // ---- K chunk k32 = 32 (consumes bnxt, issues next stage's k32=0) ----
    v16bf a1 = bfcat(*(const v8bf*)(Ab + 32 + kh * 16),
                     *(const v8bf*)(Ab + 40 + kh * 16));
    if (s + 1 < NSTAGE) {
#pragma unroll
      for (int tt = 0; tt < NT; ++tt) {
        __builtin_prefetch(bbase[tt] + koff + 2 * KC, 0, 1);   // stage s+2 line
        bcur[tt] = loadB(bbase[tt] + koff + KC);               // stage s+1, k32=0
      }
    }
#pragma unroll
    for (int tt = 0; tt < NT; ++tt)
      acc[tt] = __builtin_amdgcn_wmma_f32_16x16x32_bf16(
          false, a1, false, bnxt[tt], (short)0, acc[tt], false, false);

    koff += KC;
    __syncthreads();   // before stage s+2 prefetch overwrites buffer (s&1)
  }
}

// ---------------- weight prep ----------------

// W[K][N] fp32 -> Wt[N][K] bf16 (transpose + downconvert).
__global__ __launch_bounds__(256) void convert_wt(const float* __restrict__ W,
                                                  __bf16* __restrict__ Wt,
                                                  int K, int N) {
  int idx = blockIdx.x * 256 + threadIdx.x;
  if (idx >= K * N) return;
  int k = idx / N;
  int n = idx - k * N;
  Wt[(size_t)n * K + k] = (__bf16)W[idx];
}

__global__ __launch_bounds__(256) void init_h(const float* __restrict__ H0,
                                              float* __restrict__ Hc,
                                              __bf16* __restrict__ Hb) {
  int i = blockIdx.x * 256 + threadIdx.x;
  if (i < B_ * H_) { Hc[i] = H0[i]; Hb[i] = (__bf16)H0[i]; }
}

__global__ __launch_bounds__(256) void copy_hfinal(const float* __restrict__ Hc,
                                                   float* __restrict__ out) {
  int i = blockIdx.x * 256 + threadIdx.x;
  if (i < B_ * H_) out[i] = Hc[i];
}

// ---------------- per-timestep kernels ----------------

// r,z gates. Grid: 16 blocks x 64 cols. Waves 0-3: r (mt 0..3), waves 4-7: z.
__global__ __launch_bounds__(256) void gru_gates_rz(
    const __bf16* __restrict__ Hb, const float* __restrict__ Hc,
    const __bf16* __restrict__ r_wt, const __bf16* __restrict__ z_wt,
    const float* __restrict__ r_wxh, const float* __restrict__ z_wxh,
    const float* __restrict__ r_b, const float* __restrict__ z_b,
    const int* __restrict__ inputs, int t,
    __bf16* __restrict__ HRb, float* __restrict__ zbuf)
{
  __shared__ __align__(16) __bf16 As[2 * 64 * LDP];
  __shared__ int sIdx[B_];
  const int tid = threadIdx.x;
  if (tid < B_) sIdx[tid] = inputs[tid * T_ + t];   // inputs is [B][T]

  const int wave  = tid >> 5;
  const int lane  = tid & 31;
  const int gate  = wave >> 2;        // 0:r 1:z
  const int mt    = wave & 3;
  const int colWG = blockIdx.x * 64;

  v8f acc[4];
#pragma unroll
  for (int i = 0; i < 4; ++i) acc[i] = (v8f){0.f,0.f,0.f,0.f,0.f,0.f,0.f,0.f};

  gemm_staged<4>(Hb, gate ? z_wt : r_wt, mt, colWG, As, acc);

  const float* wxh  = gate ? z_wxh : r_wxh;
  const float* bias = gate ? z_b   : r_b;
  const int kh = lane >> 4, nl = lane & 15;
#pragma unroll
  for (int tt = 0; tt < 4; ++tt) {
    const int ncol = colWG + tt * 16 + nl;
    const float bn = bias[ncol];
#pragma unroll
    for (int i = 0; i < 8; ++i) {
      const int brow = mt * 16 + kh * 8 + i;
      const int tok  = sIdx[brow];
      const float pre = acc[tt][i] + wxh[(size_t)tok * H_ + ncol] + bn;
      const float g = 1.0f / (1.0f + __expf(-pre));
      if (gate == 0) HRb[brow * H_ + ncol] = (__bf16)(Hc[brow * H_ + ncol] * g);
      else           zbuf[brow * H_ + ncol] = g;
    }
  }
}

// h-tilde + state update. Grid: 16 blocks x 64 cols. 8 waves x 2 tiles.
__global__ __launch_bounds__(256) void gru_gate_h(
    const __bf16* __restrict__ HRb, const float* __restrict__ zbuf,
    float* __restrict__ Hc, __bf16* __restrict__ Hb,
    const __bf16* __restrict__ h_wt, const float* __restrict__ h_wxh,
    const float* __restrict__ h_b, const int* __restrict__ inputs, int t)
{
  __shared__ __align__(16) __bf16 As[2 * 64 * LDP];
  __shared__ int sIdx[B_];
  const int tid = threadIdx.x;
  if (tid < B_) sIdx[tid] = inputs[tid * T_ + t];

  const int wave  = tid >> 5;
  const int lane  = tid & 31;
  const int mt    = wave >> 1;
  const int ntb   = (wave & 1) * 2;
  const int colWG = blockIdx.x * 64;

  v8f acc[2];
#pragma unroll
  for (int i = 0; i < 2; ++i) acc[i] = (v8f){0.f,0.f,0.f,0.f,0.f,0.f,0.f,0.f};

  gemm_staged<2>(HRb, h_wt, mt, colWG + ntb * 16, As, acc);

  const int kh = lane >> 4, nl = lane & 15;
#pragma unroll
  for (int tt = 0; tt < 2; ++tt) {
    const int ncol = colWG + (ntb + tt) * 16 + nl;
    const float bn = h_b[ncol];
#pragma unroll
    for (int i = 0; i < 8; ++i) {
      const int brow = mt * 16 + kh * 8 + i;
      const int tok  = sIdx[brow];
      const float ht = tanhf(acc[tt][i] + h_wxh[(size_t)tok * H_ + ncol] + bn);
      const float z  = zbuf[brow * H_ + ncol];
      const float hn = z * Hc[brow * H_ + ncol] + (1.0f - z) * ht;
      Hc[brow * H_ + ncol] = hn;           // in-place: one thread per element
      Hb[brow * H_ + ncol] = (__bf16)hn;
    }
  }
}

// Output projection Y = Hn @ o_w + o_b. Grid: 32 blocks x 128 cols. 8 waves x 4 tiles.
__global__ __launch_bounds__(256) void gru_out(
    const __bf16* __restrict__ Hb, const __bf16* __restrict__ o_wt,
    const float* __restrict__ o_b, float* __restrict__ Y)
{
  __shared__ __align__(16) __bf16 As[2 * 64 * LDP];
  const int tid   = threadIdx.x;
  const int wave  = tid >> 5;
  const int lane  = tid & 31;
  const int mt    = wave & 3;
  const int nh    = wave >> 2;
  const int colWG = blockIdx.x * 128;

  v8f acc[4];
#pragma unroll
  for (int i = 0; i < 4; ++i) acc[i] = (v8f){0.f,0.f,0.f,0.f,0.f,0.f,0.f,0.f};

  gemm_staged<4>(Hb, o_wt, mt, colWG + nh * 64, As, acc);

  const int kh = lane >> 4, nl = lane & 15;
#pragma unroll
  for (int tt = 0; tt < 4; ++tt) {
    const int ncol = colWG + nh * 64 + tt * 16 + nl;
    const float bn = o_b[ncol];
#pragma unroll
    for (int i = 0; i < 8; ++i) {
      const int brow = mt * 16 + kh * 8 + i;
      Y[(size_t)brow * V_ + ncol] = acc[tt][i] + bn;
    }
  }
}

// ---------------- host driver ----------------

extern "C" void kernel_launch(void* const* d_in, const int* in_sizes, int n_in,
                              void* d_out, int out_size, void* d_ws, size_t ws_size,
                              hipStream_t stream) {
  const int*   inputs = (const int*)  d_in[0];
  const float* H0     = (const float*)d_in[1];
  const float* r_wxh  = (const float*)d_in[2];
  const float* r_whh  = (const float*)d_in[3];
  const float* r_b    = (const float*)d_in[4];
  const float* z_wxh  = (const float*)d_in[5];
  const float* z_whh  = (const float*)d_in[6];
  const float* z_b    = (const float*)d_in[7];
  const float* h_wxh  = (const float*)d_in[8];
  const float* h_whh  = (const float*)d_in[9];
  const float* h_b    = (const float*)d_in[10];
  const float* o_who  = (const float*)d_in[11];
  const float* o_b    = (const float*)d_in[12];

  // Workspace carve (~14.8 MB total)
  char* ws = (char*)d_ws;
  __bf16* r_wt = (__bf16*)ws; ws += (size_t)H_ * H_ * 2;
  __bf16* z_wt = (__bf16*)ws; ws += (size_t)H_ * H_ * 2;
  __bf16* h_wt = (__bf16*)ws; ws += (size_t)H_ * H_ * 2;
  __bf16* o_wt = (__bf16*)ws; ws += (size_t)H_ * V_ * 2;
  __bf16* Hb   = (__bf16*)ws; ws += (size_t)B_ * H_ * 2;
  __bf16* HRb  = (__bf16*)ws; ws += (size_t)B_ * H_ * 2;
  float*  Hc   = (float*) ws; ws += (size_t)B_ * H_ * 4;
  float*  zbuf = (float*) ws; ws += (size_t)B_ * H_ * 4;

  // Weight prep (recomputed every call; deterministic).
  const int gridHH = (H_ * H_ + 255) / 256;
  const int gridHV = (H_ * V_ + 255) / 256;
  convert_wt<<<gridHH, 256, 0, stream>>>(r_whh, r_wt, H_, H_);
  convert_wt<<<gridHH, 256, 0, stream>>>(z_whh, z_wt, H_, H_);
  convert_wt<<<gridHH, 256, 0, stream>>>(h_whh, h_wt, H_, H_);
  convert_wt<<<gridHV, 256, 0, stream>>>(o_who, o_wt, H_, V_);  // [H][V] -> [V][H]
  init_h<<<(B_ * H_ + 255) / 256, 256, 0, stream>>>(H0, Hc, Hb);

  float* out = (float*)d_out;
  for (int t = 0; t < T_; ++t) {
    gru_gates_rz<<<H_ / 64, 256, 0, stream>>>(Hb, Hc, r_wt, z_wt, r_wxh, z_wxh,
                                              r_b, z_b, inputs, t, HRb, zbuf);
    gru_gate_h<<<H_ / 64, 256, 0, stream>>>(HRb, zbuf, Hc, Hb, h_wt, h_wxh,
                                            h_b, inputs, t);
    gru_out<<<V_ / 128, 256, 0, stream>>>(Hb, o_wt, o_b, out + (size_t)t * B_ * V_);
  }
  copy_hfinal<<<(B_ * H_ + 255) / 256, 256, 0, stream>>>(
      Hc, out + (size_t)T_ * B_ * V_);
}
